// Drug2PathDifferCrossMHA_89739046682964
// MI455X (gfx1250) — matrix operand order repaired and checked by reading, compiled, not run on hardware
//
#include <hip/hip_runtime.h>
#include <math.h>

#define B_   64
#define P_   2048
#define E_   512
#define D_   1024
#define H_   8
#define HD_  64
#define HS_  16    // H * 2 (two score branches per head)
#define DH2_ 128   // 2*HD

// LAMBDA_INIT = 0.8 - 0.6*exp(-0.3)
#define LAMBDA_INIT_F  0.35550906759051307f
#define ONE_MINUS_LI_F 0.6444909324094869f
#define SCALING_F      0.125f

typedef __attribute__((ext_vector_type(16))) __bf16    v16bf;
typedef __attribute__((ext_vector_type(8)))  __bf16    v8bf;
typedef __attribute__((ext_vector_type(8)))  float     v8f;
typedef __attribute__((ext_vector_type(8)))  unsigned  v8u;

static __device__ __forceinline__ __bf16 to_bf16(float f) {
  unsigned u = __builtin_bit_cast(unsigned, f);
  unsigned r = (u + 0x7FFFu + ((u >> 16) & 1u)) >> 16;   // round-to-nearest-even
  return __builtin_bit_cast(__bf16, (unsigned short)r);
}

// Pack two f32 -> one dword holding {bf16(f0) in [15:0], bf16(f1) in [31:16]}
// via RNE bias adds + v_perm_b32 byte select (bytes {u1.3,u1.2,u0.3,u0.2}).
static __device__ __forceinline__ unsigned pack2_bf16(float f0, float f1) {
  unsigned u0 = __builtin_bit_cast(unsigned, f0);
  unsigned u1 = __builtin_bit_cast(unsigned, f1);
  u0 += 0x7FFFu + ((u0 >> 16) & 1u);
  u1 += 0x7FFFu + ((u1 >> 16) & 1u);
  return __builtin_amdgcn_perm(u1, u0, 0x07060302u);
}

// ---------------------------------------------------------------------------
// 1) lam = exp(sum(lq1*lk1)) - exp(sum(lq2*lk2)) + LAMBDA_INIT   (1 block, 64 thr)
// ---------------------------------------------------------------------------
__global__ void d2p_lambda(const float* __restrict__ lq1, const float* __restrict__ lk1,
                           const float* __restrict__ lq2, const float* __restrict__ lk2,
                           float* __restrict__ lam_out) {
  __shared__ float r1[64], r2[64];
  int t = threadIdx.x;
  r1[t] = lq1[t] * lk1[t];
  r2[t] = lq2[t] * lk2[t];
  __syncthreads();
  for (int s = 32; s > 0; s >>= 1) {
    if (t < s) { r1[t] += r1[t + s]; r2[t] += r2[t + s]; }
    __syncthreads();
  }
  if (t == 0) lam_out[0] = expf(r1[0]) - expf(r2[0]) + LAMBDA_INIT_F;
}

// ---------------------------------------------------------------------------
// 2) Per batch: Qs = (query@Wq)*SCALING;  U_T[hs][e] = sum_d Wk[e][hs*64+d]*Qs[hs*64+d]
//    Also zero pad-rows 8..15 of diffpad. (grid 64, 256 thr)
// ---------------------------------------------------------------------------
__global__ void d2p_prep(const float* __restrict__ query, const float* __restrict__ Wq,
                         const float* __restrict__ Wk, __bf16* __restrict__ U,
                         __bf16* __restrict__ diffpad) {
  __shared__ float q_l[E_];
  __shared__ float qs_l[D_];
  int b = blockIdx.x, t = threadIdx.x;
  for (int i = t; i < E_; i += 256) q_l[i] = query[(size_t)b * E_ + i];
  __syncthreads();
  for (int d = t; d < D_; d += 256) {
    float acc = 0.f;
    for (int e = 0; e < E_; ++e) acc += q_l[e] * Wq[(size_t)e * D_ + d];
    qs_l[d] = acc * SCALING_F;
  }
  __syncthreads();
  for (int idx = t; idx < HS_ * E_; idx += 256) {
    int hs = idx >> 9, e = idx & (E_ - 1);
    const float* wrow = Wk + (size_t)e * D_ + hs * HD_;
    const float* qrow = qs_l + hs * HD_;
    float acc = 0.f;
#pragma unroll 8
    for (int d = 0; d < HD_; ++d) acc += wrow[d] * qrow[d];
    U[((size_t)b * HS_ + hs) * E_ + e] = to_bf16(acc);
  }
  __bf16 z = to_bf16(0.f);
  for (int i = t; i < 8 * P_; i += 256)
    diffpad[((size_t)b * HS_ + 8) * P_ + i] = z;
}

// ---------------------------------------------------------------------------
// 3) scores[b][hs][p] = key[b,p,:] @ U_T[b][hs][:]   via v_wmma_f32_16x16x32_bf16
//    grid (64, P/256), 256 thr = 8 waves; each wave: 2 M-tiles (32 p-rows), K=512.
// ---------------------------------------------------------------------------
__global__ void d2p_scores(const float* __restrict__ key, const __bf16* __restrict__ U,
                           float* __restrict__ scores) {
  int b = blockIdx.x;
  int p0 = blockIdx.y * 256;
  int wv = threadIdx.x >> 5, lane = threadIdx.x & 31;
  int m = lane & 15, half = lane >> 4;
  int rowbase = p0 + wv * 32;

  v8f c0 = {}; v8f c1 = {};
  const __bf16* urow = U + ((size_t)b * HS_ + m) * E_;   // B-matrix column n == lane&15

  for (int k0 = 0; k0 < E_; k0 += 32) {
    v8bf blo = *(const v8bf*)(urow + k0 + 8 * half);        // 16B aligned, table k order
    v8bf bhi = *(const v8bf*)(urow + k0 + 16 + 8 * half);
    v16bf bfrag = __builtin_shufflevector(blo, bhi, 0, 1, 2, 3, 4, 5, 6, 7,
                                          8, 9, 10, 11, 12, 13, 14, 15);
    const float* a0p = key + ((size_t)b * P_ + rowbase + m) * E_ + k0;
    const float* a1p = key + ((size_t)b * P_ + rowbase + 16 + m) * E_ + k0;
    v8u p0v, p1v;
#pragma unroll
    for (int i = 0; i < 4; ++i) {
      p0v[i]     = pack2_bf16(a0p[8 * half + 2 * i],      a0p[8 * half + 2 * i + 1]);
      p0v[4 + i] = pack2_bf16(a0p[16 + 8 * half + 2 * i], a0p[16 + 8 * half + 2 * i + 1]);
      p1v[i]     = pack2_bf16(a1p[8 * half + 2 * i],      a1p[8 * half + 2 * i + 1]);
      p1v[4 + i] = pack2_bf16(a1p[16 + 8 * half + 2 * i], a1p[16 + 8 * half + 2 * i + 1]);
    }
    v16bf a0 = __builtin_bit_cast(v16bf, p0v);
    v16bf a1 = __builtin_bit_cast(v16bf, p1v);
    c0 = __builtin_amdgcn_wmma_f32_16x16x32_bf16(false, a0, false, bfrag, (short)0, c0, false, false);
    c1 = __builtin_amdgcn_wmma_f32_16x16x32_bf16(false, a1, false, bfrag, (short)0, c1, false, false);
  }
  // D layout: n = lane&15 (== hs), row = v + 8*half within tile
  float* s0 = scores + ((size_t)b * HS_ + m) * P_ + rowbase + 8 * half;
#pragma unroll
  for (int v = 0; v < 8; ++v) s0[v] = c0[v];
  float* s1 = s0 + 16;
#pragma unroll
  for (int v = 0; v < 8; ++v) s1[v] = c1[v];
}

// ---------------------------------------------------------------------------
// 4) Softmax over P for hs=2h,2h+1; diff = sm0 - lam*sm1. Writes diff_attn output
//    (f32) and diffpad (bf16, rows 0..7). grid (64,8), 256 thr, 8 elems/thr.
// ---------------------------------------------------------------------------
__global__ void d2p_softmax(const float* __restrict__ scores, const float* __restrict__ lam_p,
                            float* __restrict__ diff_out, __bf16* __restrict__ diffpad) {
  __shared__ float red[256];
  int b = blockIdx.x, h = blockIdx.y, t = threadIdx.x;
  const float lam = lam_p[0];
  const float* s0 = scores + ((size_t)b * HS_ + 2 * h) * P_;
  const float* s1 = s0 + P_;
  float v0[8], v1[8];
  float m0 = -__builtin_inff(), m1 = -__builtin_inff();
#pragma unroll
  for (int i = 0; i < 8; ++i) {
    int p = t + i * 256;
    v0[i] = s0[p]; v1[i] = s1[p];
    m0 = fmaxf(m0, v0[i]); m1 = fmaxf(m1, v1[i]);
  }
  red[t] = m0; __syncthreads();
  for (int s = 128; s > 0; s >>= 1) { if (t < s) red[t] = fmaxf(red[t], red[t + s]); __syncthreads(); }
  m0 = red[0]; __syncthreads();
  red[t] = m1; __syncthreads();
  for (int s = 128; s > 0; s >>= 1) { if (t < s) red[t] = fmaxf(red[t], red[t + s]); __syncthreads(); }
  m1 = red[0]; __syncthreads();

  float t0 = 0.f, t1 = 0.f;
#pragma unroll
  for (int i = 0; i < 8; ++i) {
    v0[i] = expf(v0[i] - m0); t0 += v0[i];
    v1[i] = expf(v1[i] - m1); t1 += v1[i];
  }
  red[t] = t0; __syncthreads();
  for (int s = 128; s > 0; s >>= 1) { if (t < s) red[t] += red[t + s]; __syncthreads(); }
  t0 = red[0]; __syncthreads();
  red[t] = t1; __syncthreads();
  for (int s = 128; s > 0; s >>= 1) { if (t < s) red[t] += red[t + s]; __syncthreads(); }
  t1 = red[0];

  float inv0 = 1.f / (t0 + 1e-20f);
  float inv1 = lam / (t1 + 1e-20f);
  float* dout = diff_out + ((size_t)b * H_ + h) * P_;
  __bf16* dpad = diffpad + ((size_t)b * HS_ + h) * P_;
#pragma unroll
  for (int i = 0; i < 8; ++i) {
    int p = t + i * 256;
    float d = v0[i] * inv0 - v1[i] * inv1;
    dout[p] = d;
    dpad[p] = to_bf16(d);
  }
}

// ---------------------------------------------------------------------------
// 5) w[b][h][e] = sum_p diff[b][h][p] * key[b][p][e]  via WMMA (A = diff padded
//    to 16 rows, B = key tile). grid (64,4), 256 thr = 8 waves; wave owns 16 e-cols.
// ---------------------------------------------------------------------------
__global__ void d2p_attnv(const float* __restrict__ key, const __bf16* __restrict__ diffpad,
                          float* __restrict__ w) {
  int b = blockIdx.x;
  int wv = threadIdx.x >> 5, lane = threadIdx.x & 31;
  int n = lane & 15, half = lane >> 4;
  int e = blockIdx.y * 128 + wv * 16 + n;
  const __bf16* arow = diffpad + ((size_t)b * HS_ + n) * P_;   // A row m == lane&15
  v8f c = {};
  for (int k0 = 0; k0 < P_; k0 += 32) {
    v16bf a = *(const v16bf*)(arow + k0 + 16 * half);          // 32B aligned
    const float* kp = key + ((size_t)b * P_ + k0 + 16 * half) * E_ + e;
    v8u pk;
#pragma unroll
    for (int j = 0; j < 8; ++j)
      pk[j] = pack2_bf16(kp[(size_t)(2 * j) * E_], kp[(size_t)(2 * j + 1) * E_]);
    v16bf bf = __builtin_bit_cast(v16bf, pk);
    c = __builtin_amdgcn_wmma_f32_16x16x32_bf16(false, a, false, bf, (short)0, c, false, false);
  }
  if (half == 0) {   // D rows 0..7 = valid heads (rows 8..15 are zero padding)
#pragma unroll
    for (int v = 0; v < 8; ++v) w[((size_t)b * H_ + v) * E_ + e] = c[v];
  }
}

// ---------------------------------------------------------------------------
// 6) out_h = w @ Wv-slice; RMSNorm per head; out = x @ Wo. grid 64, 256 thr.
// ---------------------------------------------------------------------------
__global__ void d2p_final(const float* __restrict__ w, const float* __restrict__ Wv,
                          const float* __restrict__ Wo, const float* __restrict__ rmsw,
                          float* __restrict__ out) {
  __shared__ float w_l[H_ * E_];   // 16 KB
  __shared__ float x_l[D_];        // 4 KB
  __shared__ float red[256];
  __shared__ float scl[H_];
  int b = blockIdx.x, t = threadIdx.x;
  for (int i = t; i < H_ * E_; i += 256) w_l[i] = w[(size_t)b * H_ * E_ + i];
  __syncthreads();
  for (int idx = t; idx < D_; idx += 256) {
    int h = idx >> 7;
    const float* wr = w_l + h * E_;
    float acc = 0.f;
    for (int e = 0; e < E_; ++e) acc += wr[e] * Wv[(size_t)e * D_ + idx];
    x_l[idx] = acc;
  }
  __syncthreads();
  {
    int h = t >> 5, l = t & 31;   // one wave per head
    float s = 0.f;
#pragma unroll
    for (int i = 0; i < 4; ++i) { float v = x_l[h * DH2_ + l + i * 32]; s += v * v; }
    red[t] = s; __syncthreads();
    for (int stp = 16; stp > 0; stp >>= 1) { if (l < stp) red[t] += red[t + stp]; __syncthreads(); }
    if (l == 0) scl[h] = rsqrtf(red[t] * (1.0f / 128.0f) + 1e-5f);
  }
  __syncthreads();
  for (int idx = t; idx < D_; idx += 256) {
    int h = idx >> 7;
    x_l[idx] = x_l[idx] * scl[h] * rmsw[idx & 127] * ONE_MINUS_LI_F;
  }
  __syncthreads();
  for (int o = t; o < E_; o += 256) {
    float acc = 0.f;
    for (int d = 0; d < D_; ++d) acc += x_l[d] * Wo[(size_t)d * E_ + o];
    out[(size_t)b * E_ + o] = acc;
  }
}

// ---------------------------------------------------------------------------
extern "C" void kernel_launch(void* const* d_in, const int* in_sizes, int n_in,
                              void* d_out, int out_size, void* d_ws, size_t ws_size,
                              hipStream_t stream) {
  (void)in_sizes; (void)n_in; (void)out_size; (void)ws_size;
  const float* query = (const float*)d_in[0];
  const float* key   = (const float*)d_in[1];
  const float* Wq    = (const float*)d_in[2];
  const float* Wk    = (const float*)d_in[3];
  const float* Wv    = (const float*)d_in[4];
  const float* Wo    = (const float*)d_in[5];
  const float* lq1   = (const float*)d_in[6];
  const float* lk1   = (const float*)d_in[7];
  const float* lq2   = (const float*)d_in[8];
  const float* lk2   = (const float*)d_in[9];
  const float* rmsw  = (const float*)d_in[10];

  float* out  = (float*)d_out;          // [64,512]
  float* diff = out + (size_t)B_ * E_;  // [64,8,2048]

  char* ws = (char*)d_ws;
  size_t off = 0;
  float*  lam     = (float*)(ws + off);  off += 256;
  __bf16* U       = (__bf16*)(ws + off); off += (size_t)B_ * HS_ * E_ * 2;   // 1 MB
  __bf16* diffpad = (__bf16*)(ws + off); off += (size_t)B_ * HS_ * P_ * 2;   // 4 MB
  float*  scores  = (float*)(ws + off);  off += (size_t)B_ * HS_ * P_ * 4;   // 8 MB
  float*  wbuf    = (float*)(ws + off);  off += (size_t)B_ * H_ * E_ * 4;    // 1 MB

  d2p_lambda <<<1, 64, 0, stream>>>(lq1, lk1, lq2, lk2, lam);
  d2p_prep   <<<B_, 256, 0, stream>>>(query, Wq, Wk, U, diffpad);
  d2p_scores <<<dim3(B_, P_ / 256), 256, 0, stream>>>(key, U, scores);
  d2p_softmax<<<dim3(B_, H_), 256, 0, stream>>>(scores, lam, diff, diffpad);
  d2p_attnv  <<<dim3(B_, 4), 256, 0, stream>>>(key, diffpad, wbuf);
  d2p_final  <<<B_, 256, 0, stream>>>(wbuf, Wv, Wo, rmsw, out);
}
